// SSGCModel_57208964383026
// MI455X (gfx1250) — matrix operand order repaired and checked by reading, not compiled, not run on hardware
//
#include <hip/hip_runtime.h>
#include <hip/hip_bf16.h>
#include <stdint.h>

typedef __attribute__((ext_vector_type(16))) __bf16 v16bf;
typedef __attribute__((ext_vector_type(8)))  float  v8f;

#define F_IN   128
#define HIDDEN 64
#define NCLS   40
#define KHOPS  10

// ---------- small utility kernels ----------

__global__ void k_fill(float* __restrict__ p, float v, int n) {
    int i = blockIdx.x * blockDim.x + threadIdx.x;
    if (i < n) p[i] = v;
}

__global__ void k_deg_edges(const int* __restrict__ row, const float* __restrict__ w,
                            float* __restrict__ deg, int E) {
    int e = blockIdx.x * blockDim.x + threadIdx.x;
    if (e < E) atomicAdd(&deg[row[e]], w[e]);
}

__global__ void k_dinv(float* __restrict__ deg, int n) {
    int i = blockIdx.x * blockDim.x + threadIdx.x;
    if (i < n) {
        float d = deg[i];
        deg[i] = (d > 0.f) ? rsqrtf(fmaxf(d, 1e-12f)) : 0.f;
    }
}

// ---------- SpMM: one wave (32 lanes) per edge, float4 per lane ----------

__global__ void k_spmm(const int* __restrict__ row, const int* __restrict__ col,
                       const float* __restrict__ w, const float* __restrict__ dinv,
                       const float* __restrict__ h, float* __restrict__ hn, int E) {
    int gid  = blockIdx.x * blockDim.x + threadIdx.x;
    int e    = gid >> 5;
    int lane = gid & 31;
    if (e >= E) return;
    int r = row[e], c = col[e];
    float nw = dinv[r] * w[e] * dinv[c];
    const float4 hv = *(const float4*)(h + (size_t)c * F_IN + lane * 4);
    float* dst = hn + (size_t)r * F_IN + lane * 4;
    atomicAdd(dst + 0, nw * hv.x);
    atomicAdd(dst + 1, nw * hv.y);
    atomicAdd(dst + 2, nw * hv.z);
    atomicAdd(dst + 3, nw * hv.w);
}

// self-loop term + hop accumulation, float4 granularity
__global__ void k_finalize(const float* __restrict__ dinv, const float* __restrict__ h,
                           float* __restrict__ hn, float* __restrict__ acc,
                           int first, int nquads) {
    int q = blockIdx.x * blockDim.x + threadIdx.x;
    if (q >= nquads) return;
    int node = q >> 5;                      // 128 floats = 32 quads per node
    float di = dinv[node];
    float d2 = di * di;
    float4 hv  = ((const float4*)h)[q];
    float4 hnv = ((float4*)hn)[q];
    hnv.x += d2 * hv.x; hnv.y += d2 * hv.y;
    hnv.z += d2 * hv.z; hnv.w += d2 * hv.w;
    ((float4*)hn)[q] = hnv;
    float4 a;
    if (first) {
        a = hnv;
    } else {
        a = ((const float4*)acc)[q];
        a.x += hnv.x; a.y += hnv.y; a.z += hnv.z; a.w += hnv.w;
    }
    ((float4*)acc)[q] = a;
}

// h = (1-alpha)/K * acc + alpha * x, converted to bf16 for the WMMA head
__global__ void k_mix(const float* __restrict__ acc, const float* __restrict__ x,
                      __bf16* __restrict__ hbf, int total) {
    int i = blockIdx.x * blockDim.x + threadIdx.x;
    if (i < total) hbf[i] = (__bf16)(0.09f * acc[i] + 0.1f * x[i]);
}

// ---------- weight repack into WMMA B-fragment layout ----------
// B fragment (bf16 32x16): element i of lane L -> K = 32*kt + 8*(L>=16) + (i<8 ? i : i+8), N = 16*nt + (L&15)

__global__ void k_repack_w1(const float* __restrict__ W1, __bf16* __restrict__ w1f) {
    int fi = blockIdx.x;          // fi = kt*4 + nt, 16 fragments
    int lane = threadIdx.x;       // 0..31
    int kt = fi >> 2, nt = fi & 3;
    int lrow = lane & 15, hi = lane >> 4;
#pragma unroll
    for (int i = 0; i < 16; ++i) {
        int K = kt * 32 + (i < 8 ? i : i + 8) + hi * 8;
        int N = nt * 16 + lrow;
        w1f[((size_t)fi * 32 + lane) * 16 + i] = (__bf16)W1[(size_t)K * HIDDEN + N];
    }
}

__global__ void k_repack_w2(const float* __restrict__ W2, __bf16* __restrict__ w2f) {
    int fi = blockIdx.x;          // fi = kt*3 + nt, 6 fragments (N padded 40 -> 48)
    int lane = threadIdx.x;
    int kt = fi / 3, nt = fi % 3;
    int lrow = lane & 15, hi = lane >> 4;
#pragma unroll
    for (int i = 0; i < 16; ++i) {
        int K = kt * 32 + (i < 8 ? i : i + 8) + hi * 8;
        int N = nt * 16 + lrow;
        float val = (N < NCLS) ? W2[(size_t)K * NCLS + N] : 0.f;
        w2f[((size_t)fi * 32 + lane) * 16 + i] = (__bf16)val;
    }
}

// ---------- fused MLP head: relu(H@W1+b1)@W2+b2 via v_wmma_f32_16x16x32_bf16 ----------
// 4 waves per block, one 16-row node tile per wave.
// A fragment (bf16 16x32): lane L row M = L&15; element i -> K = 32*kt + 8*(L>=16) + (i<8 ? i : i+8)

__global__ void __launch_bounds__(128) k_mlp(const __bf16* __restrict__ hbf,
                                             const __bf16* __restrict__ w1f,
                                             const float* __restrict__ b1,
                                             const __bf16* __restrict__ w2f,
                                             const float* __restrict__ b2,
                                             float* __restrict__ out,
                                             int nodes, int ntiles) {
    __shared__ __bf16 lds[4][16 * HIDDEN];
    int lane = threadIdx.x & 31;
    int wave = threadIdx.x >> 5;
    int tile = blockIdx.x * 4 + wave;
    if (tile >= ntiles) return;                 // wave-uniform: EXEC stays all-ones
    int lrow = lane & 15;
    int hi   = lane >> 4;
    int row0 = tile * 16;
    int arow = row0 + lrow;
    if (arow >= nodes) arow = nodes - 1;

    // ---- layer 1: [16x128] @ [128x64] -> 16 WMMAs ----
    v8f c1[4] = {};
#pragma unroll
    for (int kt = 0; kt < 4; ++kt) {
        const __bf16* ap = hbf + (size_t)arow * F_IN + kt * 32 + hi * 8;
        v16bf a;
#pragma unroll
        for (int i = 0; i < 8; ++i) { a[i] = ap[i]; a[8 + i] = ap[16 + i]; }
#pragma unroll
        for (int nt = 0; nt < 4; ++nt) {
            v16bf b = *(const v16bf*)(w1f + ((size_t)(kt * 4 + nt) * 32 + lane) * 16);
            c1[nt] = __builtin_amdgcn_wmma_f32_16x16x32_bf16(
                false, a, false, b, (short)0, c1[nt], false, false);
        }
    }

    // bias + relu, park layer-1 activations in LDS (row-major 16x64, per-wave region)
#pragma unroll
    for (int nt = 0; nt < 4; ++nt) {
        int ncol = nt * 16 + lrow;
        float bias = b1[ncol];
#pragma unroll
        for (int v = 0; v < 8; ++v) {
            int m = v + hi * 8;
            float val = c1[nt][v] + bias;
            val = val > 0.f ? val : 0.f;
            lds[wave][m * HIDDEN + ncol] = (__bf16)val;
        }
    }
    asm volatile("s_wait_dscnt 0x0" ::: "memory");   // intra-wave LDS RAW fence

    // ---- layer 2: [16x64] @ [64x48] -> 6 WMMAs ----
    v8f c2[3] = {};
#pragma unroll
    for (int kt = 0; kt < 2; ++kt) {
        const __bf16* lp = &lds[wave][lrow * HIDDEN + kt * 32 + hi * 8];
        v16bf a;
#pragma unroll
        for (int i = 0; i < 8; ++i) { a[i] = lp[i]; a[8 + i] = lp[16 + i]; }
#pragma unroll
        for (int nt = 0; nt < 3; ++nt) {
            v16bf b = *(const v16bf*)(w2f + ((size_t)(kt * 3 + nt) * 32 + lane) * 16);
            c2[nt] = __builtin_amdgcn_wmma_f32_16x16x32_bf16(
                false, a, false, b, (short)0, c2[nt], false, false);
        }
    }

    // bias + store logits (drop padded cols 40..47)
#pragma unroll
    for (int nt = 0; nt < 3; ++nt) {
        int ncol = nt * 16 + lrow;
        if (ncol < NCLS) {
            float bias = b2[ncol];
#pragma unroll
            for (int v = 0; v < 8; ++v) {
                int m = v + hi * 8;
                int r = row0 + m;
                if (r < nodes) out[(size_t)r * NCLS + ncol] = c2[nt][v] + bias;
            }
        }
    }
}

// ---------- host orchestration ----------

extern "C" void kernel_launch(void* const* d_in, const int* in_sizes, int n_in,
                              void* d_out, int out_size, void* d_ws, size_t ws_size,
                              hipStream_t stream) {
    const float* x  = (const float*)d_in[0];
    const int*   ei = (const int*)d_in[1];
    const float* ew = (const float*)d_in[2];
    const float* W1 = (const float*)d_in[3];
    const float* b1 = (const float*)d_in[4];
    const float* W2 = (const float*)d_in[5];
    const float* b2 = (const float*)d_in[6];

    const int N = in_sizes[0] / F_IN;
    const int E = in_sizes[2];
    const int* row = ei;        // edge_index[0]
    const int* col = ei + E;    // edge_index[1]

    // workspace carve-up (256B aligned)
    uintptr_t p = (uintptr_t)d_ws;
    auto al = [](uintptr_t q) { return (q + 255) & ~(uintptr_t)255; };
    float* hA   = (float*)al(p);
    float* hB   = (float*)al((uintptr_t)(hA + (size_t)N * F_IN));
    float* acc  = (float*)al((uintptr_t)(hB + (size_t)N * F_IN));
    float* dinv = (float*)al((uintptr_t)(acc + (size_t)N * F_IN));
    __bf16* w1f = (__bf16*)al((uintptr_t)(dinv + N));
    __bf16* w2f = (__bf16*)al((uintptr_t)(w1f + 16 * 32 * 16));
    __bf16* hbf = (__bf16*)hA;  // safe alias: after hop loop only acc & x are read

    const int T = 256;

    // degrees with self-loop (init 1.0), then dinv in-place
    k_fill<<<(N + T - 1) / T, T, 0, stream>>>(dinv, 1.0f, N);
    k_deg_edges<<<(E + T - 1) / T, T, 0, stream>>>(row, ew, dinv, E);
    k_dinv<<<(N + T - 1) / T, T, 0, stream>>>(dinv, N);

    const int total = N * F_IN;
    const int nq = total / 4;
    const float* hsrc = x;
    float* hdst = hA;
    for (int hop = 0; hop < KHOPS; ++hop) {
        k_fill<<<(total + T - 1) / T, T, 0, stream>>>(hdst, 0.f, total);
        long long work = (long long)E * 32;
        k_spmm<<<(int)((work + T - 1) / T), T, 0, stream>>>(row, col, ew, dinv, hsrc, hdst, E);
        k_finalize<<<(nq + T - 1) / T, T, 0, stream>>>(dinv, hsrc, hdst, acc,
                                                       hop == 0 ? 1 : 0, nq);
        hsrc = hdst;
        hdst = (hdst == hA) ? hB : hA;
    }

    k_mix<<<(total + T - 1) / T, T, 0, stream>>>(acc, x, hbf, total);
    k_repack_w1<<<16, 32, 0, stream>>>(W1, w1f);
    k_repack_w2<<<6, 32, 0, stream>>>(W2, w2f);

    int ntiles = (N + 15) / 16;
    k_mlp<<<(ntiles + 3) / 4, 128, 0, stream>>>(hbf, w1f, b1, w2f, b2,
                                                (float*)d_out, N, ntiles);
}
